// LocalConvolution_58823872086837
// MI455X (gfx1250) — matrix operand order repaired and checked by reading, not compiled, not run on hardware
//
#include <hip/hip_runtime.h>

typedef __attribute__((ext_vector_type(16))) __bf16 v16bf;
typedef __attribute__((ext_vector_type(8)))  __bf16 v8bf;
typedef __attribute__((ext_vector_type(8)))  float  v8f;
typedef __attribute__((ext_vector_type(4)))  float  f32x4;

#define ROWS  28
#define COLS  28
#define OCH   128
#define CIN   64
#define HH    32
#define WW    32
#define BATCH 64
#define KTOT  1600   // CIN*5*5
#define KSTEP 32
#define NSTEPS 50    // KTOT / KSTEP
#define AROW  40     // padded LDS row stride in bf16 elements (80 bytes)

// fp32x2 -> packed bf16x2 via the HW packed convert (RNE via MODE.round).
// The clang builtin is not declared for gfx1250 but the instruction exists.
__device__ __forceinline__ unsigned int packbf(float a, float b) {
  unsigned int r;
  asm("v_cvt_pk_bf16_f32 %0, %1, %2" : "=v"(r) : "v"(a), "v"(b));
  return r;
}

__global__ __launch_bounds__(256)
void LocalConvolution_58823872086837_kernel(const float* __restrict__ x,
                                            const float* __restrict__ w,
                                            float* __restrict__ out) {
  // Double-buffered bf16 panels: A = x patches [b][k], B = weights [o][k]
  __shared__ __attribute__((aligned(16))) unsigned short sA[2][BATCH * AROW];
  __shared__ __attribute__((aligned(16))) unsigned short sB[2][OCH * AROW];

  const int j    = blockIdx.x;        // 0..27
  const int i    = blockIdx.y;        // 0..27
  const int tid  = threadIdx.x;       // 0..255
  const int lane = tid & 31;
  const int wv   = tid >> 5;          // wave 0..7

  // Uniform (SGPR) bases; per-thread parts kept as 32-bit offsets so ISel
  // can select the saddr+voffset global-load form.
  const float* wbase = w + (size_t)(i * COLS + j) * (OCH * KTOT);
  const float* xu    = x + (size_t)i * WW + j;

  // ---- fill-assignments (per thread) ----
  const int oRow = tid >> 1;          // 0..127 : weight row
  const int half = tid & 1;           // which 16-float half of the 32-K chunk
  const int bRow = tid >> 2;          // 0..63  : batch row
  const unsigned e0 = (unsigned)(tid & 3) * 8u;  // 8-element chunk of 32
  const unsigned woff  = (unsigned)oRow * KTOT + (unsigned)half * 16u;
  const unsigned xoff0 = (unsigned)bRow * (CIN * HH * WW);

  // ---- compute-assignments (per wave) ----
  const int mt     = wv & 3;          // M tile 0..3  (16 rows of batch)
  const int ntBase = (wv >> 2) * 4;   // N tiles {0..3} or {4..7}
  const int nl     = lane & 15;
  const int g      = lane >> 4;       // lane group (0: lanes 0-15, 1: 16-31)

  f32x4 wreg[4];
  float xreg[8];

  auto loadPanel = [&](int k0) {
    // Weights: fold uniform k0 into the scalar base (1 SALU add / iter);
    // 16 consecutive floats per thread, non-temporal (single-use stream).
    const float* wk = wbase + k0;
#pragma unroll
    for (int q = 0; q < 4; ++q)
      wreg[q] = __builtin_nontemporal_load(
          reinterpret_cast<const f32x4*>(wk + woff) + q);
    // x patches: gather 8 elements, k -> (c,u,v); u32 offset from SGPR base
#pragma unroll
    for (int e = 0; e < 8; ++e) {
      unsigned kk = (unsigned)k0 + e0 + (unsigned)e;
      unsigned c  = kk / 25u;
      unsigned r  = kk - c * 25u;
      unsigned u  = r / 5u;
      unsigned v  = r - u * 5u;
      xreg[e] = xu[xoff0 + c * (HH * WW) + u * WW + v];
    }
  };

  auto storePanel = [&](int buf) {
    unsigned int pw[8];
#pragma unroll
    for (int q = 0; q < 4; ++q) {
      pw[q * 2]     = packbf(wreg[q][0], wreg[q][1]);
      pw[q * 2 + 1] = packbf(wreg[q][2], wreg[q][3]);
    }
    uint4* bd = reinterpret_cast<uint4*>(&sB[buf][oRow * AROW + half * 16]);
    bd[0] = make_uint4(pw[0], pw[1], pw[2], pw[3]);
    bd[1] = make_uint4(pw[4], pw[5], pw[6], pw[7]);
    unsigned int px[4];
#pragma unroll
    for (int q = 0; q < 4; ++q) px[q] = packbf(xreg[2 * q], xreg[2 * q + 1]);
    uint4* ad = reinterpret_cast<uint4*>(&sA[buf][bRow * AROW + e0]);
    ad[0] = make_uint4(px[0], px[1], px[2], px[3]);
  };

  v8f acc[4] = {};   // 4 N-tiles of 16x16 f32

  loadPanel(0);
  storePanel(0);
  __syncthreads();

  for (int s = 0; s < NSTEPS; ++s) {
    const int cur   = s & 1;
    const bool more = (s + 1) < NSTEPS;
    if (more) loadPanel((s + 1) * KSTEP);   // issue NT global loads early
    __builtin_amdgcn_sched_barrier(0);      // keep loads above the WMMA block

    // A fragment (16x32 bf16): lane m=mt*16+nl holds K = g*8..+7 and 16+g*8..
    const unsigned short* Ar = &sA[cur][(mt * 16 + nl) * AROW];
    v8bf alo = *reinterpret_cast<const v8bf*>(Ar + g * 8);
    v8bf ahi = *reinterpret_cast<const v8bf*>(Ar + 16 + g * 8);
    v16bf afrag = __builtin_shufflevector(alo, ahi,
        0, 1, 2, 3, 4, 5, 6, 7, 8, 9, 10, 11, 12, 13, 14, 15);

    v16bf bfrag[4];
#pragma unroll
    for (int t = 0; t < 4; ++t) {
      const unsigned short* Br =
          &sB[cur][((ntBase + t) * 16 + nl) * AROW + g * 8];
      v8bf blo = *reinterpret_cast<const v8bf*>(Br);
      v8bf bhi = *reinterpret_cast<const v8bf*>(Br + 16);
      bfrag[t] = __builtin_shufflevector(blo, bhi,
          0, 1, 2, 3, 4, 5, 6, 7, 8, 9, 10, 11, 12, 13, 14, 15);
    }
#pragma unroll
    for (int t = 0; t < 4; ++t) {
      acc[t] = __builtin_amdgcn_wmma_f32_16x16x32_bf16(
          false, afrag, false, bfrag[t], (short)0, acc[t], false, false);
    }

    __builtin_amdgcn_sched_barrier(0);      // keep cvt/stores below the WMMAs
    __syncthreads();                        // all waves done reading buf[cur]
    if (more) {
      storePanel((s + 1) & 1);              // wait_loadcnt lands here, covered
      __syncthreads();                      // panel visible before next compute
    }
  }

  // Writeback: out[b][o][i][j]; C/D layout: VGPR r, lane -> m = g*8+r, n = nl
  float* op = out + (size_t)(i * COLS + j);
#pragma unroll
  for (int t = 0; t < 4; ++t) {
    const int o = (ntBase + t) * 16 + nl;
#pragma unroll
    for (int r = 0; r < 8; ++r) {
      const int b = mt * 16 + g * 8 + r;
      op[((size_t)b * OCH + o) * (ROWS * COLS)] = acc[t][r];
    }
  }
}

extern "C" void kernel_launch(void* const* d_in, const int* in_sizes, int n_in,
                              void* d_out, int out_size, void* d_ws, size_t ws_size,
                              hipStream_t stream) {
  const float* x = (const float*)d_in[0];   // (64,64,32,32) fp32
  const float* w = (const float*)d_in[1];   // (28,28,128,64,5,5) fp32
  float* out = (float*)d_out;               // (64,128,28,28) fp32
  (void)in_sizes; (void)n_in; (void)out_size; (void)d_ws; (void)ws_size;
  dim3 grid(COLS, ROWS);
  LocalConvolution_58823872086837_kernel<<<grid, 256, 0, stream>>>(x, w, out);
}
